// RGCNModel_67714454388970
// MI455X (gfx1250) — matrix operand order repaired and checked by reading, not compiled, not run on hardware
//
#include <hip/hip_runtime.h>
#include <hip/hip_bf16.h>

// ---------------- problem constants ----------------
#define N_ENC   50000
#define N_CODE  100000
#define N_TOT   150000
#define F_IN    64
#define DIM     128
#define R_REL   4
#define B_BAS   8
#define L_LAY   2
#define E_EDG   1500000

typedef __attribute__((ext_vector_type(16))) __bf16 v16bf;
typedef __attribute__((ext_vector_type(8)))  float  v8f;

union Frag16 {           // assemble a 32-byte WMMA operand from two 16B loads
  uint4 u[2];
  v16bf v;
};

// =====================================================================
// WMMA bf16 GEMM:  C[M,128] = A[M,K](bf16) * Bt[K,128] (+ bias)
//   Bt stored transposed: Bt[col*K + k]   (bf16)
//   block (32,8): wave w owns M-tile (blockIdx.x*8 + w) and accumulates all
//   8 N-tiles; B (<=32KB) is staged into LDS once per block.
//   M multiple of 16; K compile-time (64 or 128) -> fully unrolled K loop.
// =====================================================================
template <int K>
__global__ __launch_bounds__(256)
void wmma_gemm(const __bf16* __restrict__ A, const __bf16* __restrict__ Bt,
               const float* __restrict__ bias, float* __restrict__ C, int M) {
  __shared__ __bf16 Bs[DIM * K];                 // 32 KB max (K=128)

  const int lane = threadIdx.x;                  // 0..31
  const int wv   = threadIdx.y;                  // 0..7
  const int tid  = wv * 32 + lane;

  // ---- stage Bt[0 .. 128*K) into LDS (16B chunks) ----
  constexpr int n16 = (DIM * K * 2) / 16;        // K=128 -> 2048, K=64 -> 1024
#pragma unroll
  for (int i = 0; i < n16 / 256; ++i)
    ((uint4*)Bs)[tid + i * 256] = ((const uint4*)Bt)[tid + i * 256];
  __syncthreads();

  const int mt = blockIdx.x * 8 + wv;            // M tile for this wave
  if (mt * 16 >= M) return;                      // wave-uniform exit (post-barrier)

  const int rc = lane & 15;                      // row (A) / col (B,D)
  const int kh = lane >> 4;                      // K-half selector

  v8f c[8];
#pragma unroll
  for (int nt = 0; nt < 8; ++nt) {
    const float bv = bias ? bias[nt * 16 + rc] : 0.0f;
#pragma unroll
    for (int i = 0; i < 8; ++i) c[nt][i] = bv;
  }

  const __bf16* arow = A + (size_t)(mt * 16 + rc) * K;

  // ---- preload all A fragments (K/32 of them) ----
  Frag16 a[K / 32];
#pragma unroll
  for (int ks = 0; ks < K / 32; ++ks) {
    a[ks].u[0] = *(const uint4*)(arow + ks * 32 + kh * 8);
    a[ks].u[1] = *(const uint4*)(arow + ks * 32 + 16 + kh * 8);
  }

#pragma unroll
  for (int ks = 0; ks < K / 32; ++ks) {
#pragma unroll
    for (int nt = 0; nt < 8; ++nt) {
      // B fragment from LDS: elem j<8 -> K = ks*32 + kh*8 + j, j>=8 -> K = ks*32 + 16 + kh*8 + (j-8)
      Frag16 b;
      const __bf16* bp = Bs + (nt * 16 + rc) * K + ks * 32 + kh * 8;
      b.u[0] = *(const uint4*)(bp);
      b.u[1] = *(const uint4*)(bp + 16);
      c[nt] = __builtin_amdgcn_wmma_f32_16x16x32_bf16(false, a[ks].v, false, b.v,
                                                      (short)0, c[nt],
                                                      false, false);
    }
  }

  const int wrow = mt * 16 + kh * 8;
#pragma unroll
  for (int nt = 0; nt < 8; ++nt) {
    const int wcol = nt * 16 + rc;
#pragma unroll
    for (int i = 0; i < 8; ++i)
      C[(size_t)(wrow + i) * DIM + wcol] = c[nt][i];
  }
}

// ---------------- small prep / utility kernels ----------------
__global__ __launch_bounds__(256)
void zero_i32(int* p, int n) {
  int i = blockIdx.x * 256 + threadIdx.x;
  if (i < n) p[i] = 0;
}

// fp32 -> bf16 bulk convert (4 elems/thread, n multiple of 4)
__global__ __launch_bounds__(256)
void convert_bf16(const float* __restrict__ x, __bf16* __restrict__ y, int n4) {
  int i = blockIdx.x * 256 + threadIdx.x;
  if (i >= n4) return;
  float4 v = ((const float4*)x)[i];
  union { __bf16 b[4]; uint2 u; } t;
  t.b[0] = (__bf16)v.x; t.b[1] = (__bf16)v.y;
  t.b[2] = (__bf16)v.z; t.b[3] = (__bf16)v.w;
  ((uint2*)y)[i] = t.u;
}

// Transpose fp32 W[Krows,128] -> bf16 Wt[128][Krows]
__global__ __launch_bounds__(256)
void transpose_bf(const float* __restrict__ W, __bf16* __restrict__ Wt, int Krows) {
  int idx = blockIdx.x * 256 + threadIdx.x;     // Krows*128 threads, exact
  int i = idx >> 7, o = idx & 127;
  Wt[(size_t)o * Krows + i] = (__bf16)W[(size_t)i * DIM + o];
}

// W_r = sum_b comp[l,r,b] * bases[l,b,:,:]  -> transposed bf16 Wt[r][o][i]
__global__ __launch_bounds__(256)
void wprep(const float* __restrict__ comp, const float* __restrict__ bases,
           __bf16* __restrict__ Wt, int l) {
  int idx = blockIdx.x * 256 + threadIdx.x;     // R*128*128 = 65536 threads, exact
  int r = idx >> 14, rem = idx & 16383;
  int i = rem >> 7, o = rem & 127;
  const float* cl = comp + (size_t)(l * R_REL + r) * B_BAS;
  const float* bl = bases + (size_t)l * B_BAS * DIM * DIM;
  float s = 0.f;
#pragma unroll
  for (int b = 0; b < B_BAS; ++b)
    s += cl[b] * bl[((size_t)b * DIM + i) * DIM + o];
  Wt[((size_t)r * DIM + o) * DIM + i] = (__bf16)s;
}

// X[N_ENC + n, :] = embed[x_code[n], :]
__global__ __launch_bounds__(256)
void embed_gather(const int* __restrict__ code, const float* __restrict__ embed,
                  float* __restrict__ X) {
  int n = blockIdx.x * 8 + threadIdx.y;
  if (n >= N_CODE) return;
  int lane = threadIdx.x;
  const float4* s = (const float4*)(embed + (size_t)code[n] * DIM);
  float4* d = (float4*)(X + (size_t)(N_ENC + n) * DIM);
  d[lane] = s[lane];
}

// per-(dst,rel) in-degree count
__global__ __launch_bounds__(256)
void count_kernel(const int* __restrict__ dst, const int* __restrict__ et,
                  int* __restrict__ cnt) {
  int e = blockIdx.x * 256 + threadIdx.x;
  if (e < E_EDG) atomicAdd(&cnt[(size_t)dst[e] * R_REL + et[e]], 1);
}

__global__ __launch_bounds__(256)
void norm_kernel(const int* __restrict__ dst, const int* __restrict__ et,
                 const int* __restrict__ cnt, float* __restrict__ norm) {
  int e = blockIdx.x * 256 + threadIdx.x;
  if (e < E_EDG) {
    int c = cnt[(size_t)dst[e] * R_REL + et[e]];
    norm[e] = 1.0f / (float)(c < 1 ? 1 : c);
  }
}

// scatter: Xn[dst] += norm * XW[src] for edges of relation r
__global__ __launch_bounds__(256)
void scatter_rel(const int* __restrict__ src, const int* __restrict__ dst,
                 const int* __restrict__ et, const float* __restrict__ norm,
                 const float* __restrict__ XW, float* __restrict__ Xn, int r) {
  int e = blockIdx.x * 8 + threadIdx.y;
  if (e >= E_EDG) return;
  if (et[e] != r) return;
  int lane = threadIdx.x;
  float nv = norm[e];
  float4 v = ((const float4*)(XW + (size_t)src[e] * DIM))[lane];
  float* d = Xn + (size_t)dst[e] * DIM + lane * 4;
  atomicAdd(d + 0, v.x * nv);
  atomicAdd(d + 1, v.y * nv);
  atomicAdd(d + 2, v.z * nv);
  atomicAdd(d + 3, v.w * nv);
}

__global__ __launch_bounds__(256)
void relu_copy(const float* __restrict__ Xn, float* __restrict__ X, int n) {
  int i = blockIdx.x * 256 + threadIdx.x;
  if (i < n) X[i] = fmaxf(Xn[i], 0.0f);
}

// logit[n] = dot(X[n,:], cls_W) + cls_b
__global__ __launch_bounds__(256)
void classifier(const float* __restrict__ X, const float* __restrict__ W,
                const float* __restrict__ b, float* __restrict__ out) {
  int row = blockIdx.x * 256 + threadIdx.x;
  if (row >= N_ENC) return;
  const float4* x4 = (const float4*)(X + (size_t)row * DIM);
  const float4* w4 = (const float4*)W;
  float s = 0.f;
#pragma unroll
  for (int i = 0; i < 32; ++i) {
    float4 a = x4[i], w = w4[i];
    s += a.x * w.x + a.y * w.y + a.z * w.z + a.w * w.w;
  }
  out[row] = s + b[0];
}

// =====================================================================
extern "C" void kernel_launch(void* const* d_in, const int* in_sizes, int n_in,
                              void* d_out, int out_size, void* d_ws, size_t ws_size,
                              hipStream_t stream) {
  const float* x_enc  = (const float*)d_in[0];
  const int*   x_code = (const int*)  d_in[1];
  const int*   e_idx  = (const int*)  d_in[2];   // [2,E]
  const int*   e_typ  = (const int*)  d_in[3];
  const float* proj_W = (const float*)d_in[4];
  const float* proj_b = (const float*)d_in[5];
  const float* embed  = (const float*)d_in[6];
  const float* comp   = (const float*)d_in[7];
  const float* bases  = (const float*)d_in[8];
  const float* root   = (const float*)d_in[9];
  const float* bias   = (const float*)d_in[10];
  const float* cls_W  = (const float*)d_in[11];
  const float* cls_b  = (const float*)d_in[12];
  float* out = (float*)d_out;

  const int* e_src = e_idx;
  const int* e_dst = e_idx + E_EDG;

  // ---- workspace layout (256B aligned slices) ----
  char*  ws  = (char*)d_ws;
  size_t off = 0;
  auto carve = [&](size_t bytes) -> void* {
    void* p = ws + off;
    off += (bytes + 255) & ~(size_t)255;
    return p;
  };
  float*  X     = (float*)carve((size_t)N_TOT * DIM * 4);
  float*  Xn    = (float*)carve((size_t)N_TOT * DIM * 4);
  float*  XW    = (float*)carve((size_t)N_TOT * DIM * 4);
  __bf16* Xbf   = (__bf16*)carve((size_t)N_TOT * DIM * 2);
  __bf16* Aebf  = (__bf16*)carve((size_t)N_ENC * F_IN * 2);
  float*  norm  = (float*)carve((size_t)E_EDG * 4);
  int*    cnt   = (int*)  carve((size_t)N_TOT * R_REL * 4);
  __bf16* Wt    = (__bf16*)carve((size_t)R_REL * DIM * DIM * 2);
  __bf16* roott = (__bf16*)carve((size_t)DIM * DIM * 2);
  __bf16* projt = (__bf16*)carve((size_t)F_IN * DIM * 2);

  dim3 blk2d(32, 8);
  const int gProj = (N_ENC / 16 + 7) / 8;        // 391
  const int gFull = (N_TOT / 16 + 7) / 8;        // 1172

  // ---- input projection + embedding gather -> X ----
  convert_bf16<<<(N_ENC * F_IN / 4 + 255) / 256, 256, 0, stream>>>(
      x_enc, Aebf, N_ENC * F_IN / 4);
  transpose_bf<<<(F_IN * DIM) / 256, 256, 0, stream>>>(proj_W, projt, F_IN);
  wmma_gemm<F_IN><<<gProj, blk2d, 0, stream>>>(Aebf, projt, proj_b, X, N_ENC);
  embed_gather<<<(N_CODE + 7) / 8, blk2d, 0, stream>>>(x_code, embed, X);

  // ---- mean-normalization coefficients ----
  const int nr = N_TOT * R_REL;
  zero_i32<<<(nr + 255) / 256, 256, 0, stream>>>(cnt, nr);
  count_kernel<<<(E_EDG + 255) / 256, 256, 0, stream>>>(e_dst, e_typ, cnt);
  norm_kernel<<<(E_EDG + 255) / 256, 256, 0, stream>>>(e_dst, e_typ, cnt, norm);

  // ---- RGCN layers ----
  for (int l = 0; l < L_LAY; ++l) {
    wprep<<<(R_REL * DIM * DIM) / 256, 256, 0, stream>>>(comp, bases, Wt, l);
    transpose_bf<<<(DIM * DIM) / 256, 256, 0, stream>>>(root + (size_t)l * DIM * DIM,
                                                        roott, DIM);
    convert_bf16<<<(N_TOT * DIM / 4 + 255) / 256, 256, 0, stream>>>(
        X, Xbf, N_TOT * DIM / 4);
    // Xn = X @ root + bias   (fully written -> no pre-zero needed)
    wmma_gemm<DIM><<<gFull, blk2d, 0, stream>>>(Xbf, roott, bias + (size_t)l * DIM,
                                                Xn, N_TOT);
    for (int r = 0; r < R_REL; ++r) {
      wmma_gemm<DIM><<<gFull, blk2d, 0, stream>>>(Xbf, Wt + (size_t)r * DIM * DIM,
                                                  nullptr, XW, N_TOT);
      scatter_rel<<<(E_EDG + 7) / 8, blk2d, 0, stream>>>(e_src, e_dst, e_typ, norm,
                                                         XW, Xn, r);
    }
    relu_copy<<<(N_TOT * DIM) / 256, 256, 0, stream>>>(Xn, X, N_TOT * DIM);
  }

  // ---- classifier ----
  classifier<<<(N_ENC + 255) / 256, 256, 0, stream>>>(X, cls_W, cls_b, out);
}
